// CollocationPhysicsLoss_43009802502473
// MI455X (gfx1250) — matrix operand order repaired and checked by reading, not compiled, not run on hardware
//
#include <hip/hip_runtime.h>
#include <hip/hip_bf16.h>

typedef __attribute__((ext_vector_type(16))) _Float16 v16h;
typedef __attribute__((ext_vector_type(8)))  _Float16 v8h;
typedef __attribute__((ext_vector_type(8)))  float    v8f;
typedef __attribute__((ext_vector_type(4)))  unsigned int u32x4;
typedef __attribute__((ext_vector_type(8)))  unsigned int u32x8;

#define N_PTS   65536
#define HID     256
#define PTSB    32            // points per block
#define W0S     30.0f
#define RHO0    1.225f
#define CSND    343.0f

// ---- LDS layout (bytes) ----
#define ACT_HALVES (5 * PTSB * HID)          // 40960 halves = 81920 B per buffer
#define OFF_ACTA   0
#define OFF_ACTB   (ACT_HALVES * 2)          // 81920
#define OFF_W      (OFF_ACTB + ACT_HALVES*2) // 163840
#define OFF_J      (OFF_W + HID*HID*2)       // 294912
#define OFF_SUMS   (OFF_J + PTSB*16*4)       // 296960
#define SMEM_BYTES (OFF_SUMS + 16)           // 296976  (<320KB WGP limit)

// ---------------------------------------------------------------
// TDM: DMA a 128KB (f16 256x256) weight tile from global to LDS.
// 1-D descriptor: data_size=8B, tensor_dim0=tile_dim0=16384 units.
// Issued per-wave (caller guards to wave 0); tracked by TENSORcnt.
// ---------------------------------------------------------------
__device__ __forceinline__ void tdm_load_w(const _Float16* gsrc, unsigned int lds_off,
                                           unsigned int units8B) {
    unsigned long long ga = (unsigned long long)(uintptr_t)gsrc;
    u32x4 g0;
    g0.x = 1u;                                   // count=1, user descriptor
    g0.y = lds_off;                              // lds_addr (bytes)
    g0.z = (unsigned int)ga;                     // global_addr[31:0]
    g0.w = (unsigned int)((ga >> 32) & 0x01FFFFFFull) | 0x80000000u; // [56:32] | type=2
    u32x8 g1;
    g1[0] = 3u << 16;                            // wg_mask=0, data_size=3 (8B), no flags
    g1[1] = (units8B & 0xFFFFu) << 16;           // abar_addr=0, tensor_dim0[15:0]
    g1[2] = ((units8B >> 16) & 0xFFFFu) | (1u << 16); // tensor_dim0[31:16], tensor_dim1=1
    g1[3] = (units8B & 0xFFFFu) << 16;           // tensor_dim1 hi=0, tile_dim0
    g1[4] = 0u;                                  // tile_dim1=0, tile_dim2=0 (unused)
    g1[5] = units8B;                             // tensor_dim0_stride[31:0]
    g1[6] = 0u;                                  // stride hi, tensor_dim1_stride lo
    g1[7] = 0u;
    asm volatile("tensor_load_to_lds %0, %1" :: "s"(g0), "s"(g1) : "memory");
}

// ---------------------------------------------------------------
// prep: room_max (mean over 8 rows, clamped 0.1) + zero loss sums
// ---------------------------------------------------------------
__global__ void prep_kernel(const float* __restrict__ room_dims, float* __restrict__ wsf) {
    int t = threadIdx.x;
    if (t < 3) {
        float s = 0.f;
        for (int i = 0; i < 8; ++i) s += room_dims[i * 3 + t];
        wsf[t] = fmaxf(s * 0.125f, 0.1f);
    }
    if (t == 4 || t == 5) wsf[t] = 0.f;   // cont / mom squared sums
}

// ---------------------------------------------------------------
// weight prep: Wt[n][k] = (f16) W[k][n] for W1,W2 ; padded Wt3 (256x16)
// ---------------------------------------------------------------
__global__ void wtrans_kernel(const float* __restrict__ W1, const float* __restrict__ W2,
                              const float* __restrict__ W3,
                              _Float16* __restrict__ wt1, _Float16* __restrict__ wt2,
                              _Float16* __restrict__ wt3) {
    int e = blockIdx.x * 256 + threadIdx.x;   // e = k*256 + n (row-major read)
    int n = e & 255;
    int k = e >> 8;
    wt1[n * 256 + k] = (_Float16)W1[e];
    wt2[n * 256 + k] = (_Float16)W2[e];
    if (e < 16 * 256) {                       // Wt3[n][k] = W3[k][n], cols 4..15 zero
        int n3 = e >> 8;
        int k3 = e & 255;
        wt3[e] = (n3 < 4) ? (_Float16)W3[k3 * 4 + n3] : (_Float16)0.f;
    }
}

// ---------------------------------------------------------------
// fused SIREN + forward-mode Jacobian + physics residual
// ---------------------------------------------------------------
__global__ __launch_bounds__(256)
void siren_pinn_kernel(const float* __restrict__ coords,
                       const float* __restrict__ time_raw,
                       const float* __restrict__ W0,
                       const float* __restrict__ b0,
                       const float* __restrict__ b1,
                       const float* __restrict__ b2,
                       const _Float16* __restrict__ wt1,
                       const _Float16* __restrict__ wt2,
                       const _Float16* __restrict__ wt3,
                       const float* __restrict__ wsf,
                       float* __restrict__ gsums) {
    extern __shared__ __align__(16) char smem[];
    _Float16* actA = (_Float16*)(smem + OFF_ACTA);
    _Float16* actB = (_Float16*)(smem + OFF_ACTB);
    _Float16* sW   = (_Float16*)(smem + OFF_W);
    float*    sJ   = (float*)(smem + OFF_J);
    float*    sSum = (float*)(smem + OFF_SUMS);

    const int tid  = threadIdx.x;
    const int lane = tid & 31;
    const int wave = tid >> 5;
    const int p0   = blockIdx.x * PTSB;
    const unsigned int sWoff = (unsigned int)(uintptr_t)sW;   // LDS byte offset

    if (tid < 2) sSum[tid] = 0.f;

    // kick off W1 DMA into LDS; it overlaps the layer-0 elementwise work below
    if (wave == 0) {
        tdm_load_w(wt1, sWoff, (HID * HID * 2) / 8);
        __builtin_prefetch((const void*)wt2, 0, 1);   // warm W2 (global_prefetch_b8)
    }

    const float rm0 = wsf[0], rm1 = wsf[1], rm2 = wsf[2];

    // ---------------- layer 0 : x(4) -> 256, plus constant-direction tangents ----
    for (int idx = tid; idx < PTSB * HID; idx += 256) {
        int n = idx >> 8;
        int j = idx & 255;
        int p = p0 + n;
        float x0 = coords[p * 3 + 0] * rm0;
        float x1 = coords[p * 3 + 1] * rm1;
        float x2 = coords[p * 3 + 2] * rm2;
        float x3 = time_raw[p] * 2.0f;
        float w0j = W0[j], w1j = W0[256 + j], w2j = W0[512 + j], w3j = W0[768 + j];
        float z = W0S * (x0 * w0j + x1 * w1j + x2 * w2j + x3 * w3j + b0[j]);
        float h = __sinf(z);
        float c = __cosf(z);
        actA[0 * 8192 + n * 256 + j] = (_Float16)h;
        actA[1 * 8192 + n * 256 + j] = (_Float16)(c * W0S * w0j);
        actA[2 * 8192 + n * 256 + j] = (_Float16)(c * W0S * w1j);
        actA[3 * 8192 + n * 256 + j] = (_Float16)(c * W0S * w2j);
        actA[4 * 8192 + n * 256 + j] = (_Float16)(c * W0S * w3j);
    }
    if (wave == 0) __builtin_amdgcn_s_wait_tensorcnt(0);   // W1 resident
    __syncthreads();

    // wave work assignment: 2 row-tiles x 16 col-tiles x 5 streams
    const int r       = wave & 1;               // row-tile (16 rows)
    const int cg      = wave >> 1;              // 4 col-tiles per wave, 2 per pass
    const int colLane = lane & 15;
    const int rowbase = r * 16 + ((lane & 16) ? 8 : 0);  // C-fragment row base
    const int aRow    = r * 16 + colLane;                // A-fragment row for this lane
    const int aKsel   = (lane & 16) ? 8 : 0;             // A K-halves select
    const int bKsel   = (lane & 16) ? 8 : 0;             // B K-halves select

    // ---------------- two hidden 256x256 layers through WMMA ----------------
    for (int layer = 0; layer < 2; ++layer) {
        const float*    bvec = layer ? b2 : b1;
        const _Float16* in   = layer ? actB : actA;
        _Float16*       out  = layer ? actA : actB;

        for (int pass = 0; pass < 2; ++pass) {
            const int nt0 = cg * 4 + pass * 2;   // first of two col-tiles
            v8f acc[5][2];
            const v8f vzero = {0.f, 0.f, 0.f, 0.f, 0.f, 0.f, 0.f, 0.f};
#pragma unroll
            for (int s = 0; s < 5; ++s) { acc[s][0] = vzero; acc[s][1] = vzero; }

#pragma unroll
            for (int kt = 0; kt < 8; ++kt) {
                // A fragments (one per stream) from activation LDS, row-major
                v16h a[5];
                const int aOff = kt * 32 + aKsel;
#pragma unroll
                for (int s = 0; s < 5; ++s) {
                    const _Float16* ap = in + s * 8192 + aRow * 256 + aOff;
                    v8h lo = *(const v8h*)ap;          // K 0..7  (or 8..15)
                    v8h hi = *(const v8h*)(ap + 16);   // K 16..23 (or 24..31)
                    a[s] = __builtin_shufflevector(lo, hi, 0,1,2,3,4,5,6,7,
                                                           8,9,10,11,12,13,14,15);
                }
                // B fragments (two col-tiles) from W^T LDS
                v16h b[2];
                const int bK = kt * 32 + bKsel;
#pragma unroll
                for (int c = 0; c < 2; ++c) {
                    const _Float16* bp = sW + ((nt0 + c) * 16 + colLane) * 256 + bK;
                    v8h lo = *(const v8h*)bp;
                    v8h hi = *(const v8h*)(bp + 16);
                    b[c] = __builtin_shufflevector(lo, hi, 0,1,2,3,4,5,6,7,
                                                           8,9,10,11,12,13,14,15);
                }
#pragma unroll
                for (int s = 0; s < 5; ++s) {
                    acc[s][0] = __builtin_amdgcn_wmma_f32_16x16x32_f16(
                        false, a[s], false, b[0], (short)0, acc[s][0], false, false);
                    acc[s][1] = __builtin_amdgcn_wmma_f32_16x16x32_f16(
                        false, a[s], false, b[1], (short)0, acc[s][1], false, false);
                }
            }

            // epilogue: z = h_in@W + b ; h = sin(z); dh_k = cos(z) * (dh_in_k@W)
#pragma unroll
            for (int c = 0; c < 2; ++c) {
                const int col  = (nt0 + c) * 16 + colLane;
                const float bb = bvec[col];
                float hv[8], cv[8];
#pragma unroll
                for (int e = 0; e < 8; ++e) {
                    float z = acc[0][c][e] + bb;
                    hv[e] = __sinf(z);
                    cv[e] = __cosf(z);
                }
#pragma unroll
                for (int e = 0; e < 8; ++e)
                    out[0 * 8192 + (rowbase + e) * 256 + col] = (_Float16)hv[e];
#pragma unroll
                for (int s = 1; s < 5; ++s)
#pragma unroll
                    for (int e = 0; e < 8; ++e)
                        out[s * 8192 + (rowbase + e) * 256 + col] =
                            (_Float16)(cv[e] * acc[s][c][e]);
            }
        }
        __syncthreads();                     // all waves done reading sW / writing out

        if (layer == 0) {                    // DMA W2 over W1
            if (wave == 0) {
                tdm_load_w(wt2, sWoff, (HID * HID * 2) / 8);
                __builtin_amdgcn_s_wait_tensorcnt(0);
            }
            __syncthreads();
        }
    }
    // final tangents dh2_k now reside in actA streams 1..4

    // ---------------- output layer via WMMA: J tiles = dh2_k @ Wt3(256x16) ----
    {   // stage padded Wt3 (8KB) into the front of the sW region
        const uint4* s3 = (const uint4*)wt3;
        uint4* d3 = (uint4*)sW;
        for (int i = tid; i < (16 * HID * 2) / 16; i += 256) d3[i] = s3[i];
    }
    __syncthreads();
    {
        const int kdir = (wave >> 1) & 3;        // tangent direction 0..3
        v8f accJ = {0.f, 0.f, 0.f, 0.f, 0.f, 0.f, 0.f, 0.f};
#pragma unroll
        for (int kt = 0; kt < 8; ++kt) {
            const _Float16* ap = actA + (1 + kdir) * 8192 + aRow * 256 + kt * 32 + aKsel;
            v8h alo = *(const v8h*)ap;
            v8h ahi = *(const v8h*)(ap + 16);
            v16h a = __builtin_shufflevector(alo, ahi, 0,1,2,3,4,5,6,7,
                                                       8,9,10,11,12,13,14,15);
            const _Float16* bp = sW + colLane * 256 + kt * 32 + bKsel;
            v8h blo = *(const v8h*)bp;
            v8h bhi = *(const v8h*)(bp + 16);
            v16h b = __builtin_shufflevector(blo, bhi, 0,1,2,3,4,5,6,7,
                                                       8,9,10,11,12,13,14,15);
            accJ = __builtin_amdgcn_wmma_f32_16x16x32_f16(
                false, a, false, b, (short)0, accJ, false, false);
        }
        if (colLane < 4) {                       // only 4 real output columns
#pragma unroll
            for (int e = 0; e < 8; ++e) {
                int p = rowbase + e;             // point row within block
                sJ[p * 16 + colLane * 4 + kdir] = accJ[e];
            }
        }
    }
    __syncthreads();

    // ---------------- physics residuals + block reduction ----------------
    if (tid < PTSB) {
        const float* J = sJ + tid * 16;          // J[i*4+k]
        float dp_dt = J[0 * 4 + 3];
        float divu  = J[1 * 4 + 0] + J[2 * 4 + 1] + J[3 * 4 + 2];
        float rc    = dp_dt + RHO0 * CSND * CSND * divu;
        float du    = J[1 * 4 + 3] + J[2 * 4 + 3] + J[3 * 4 + 3];
        float base  = RHO0 * du;
        float m0 = base + J[0], m1 = base + J[1], m2 = base + J[2];
        atomicAdd(&sSum[0], rc * rc);
        atomicAdd(&sSum[1], m0 * m0 + m1 * m1 + m2 * m2);
    }
    __syncthreads();
    if (tid == 0) {
        atomicAdd(&gsums[0], sSum[0]);
        atomicAdd(&gsums[1], sSum[1]);
    }
}

// ---------------------------------------------------------------
// finish: loss = 0.01*mean(rc^2) + 0.01*mean(rm^2)
// ---------------------------------------------------------------
__global__ void loss_finish_kernel(const float* __restrict__ gsums, float* __restrict__ out) {
    if (threadIdx.x == 0 && blockIdx.x == 0)
        out[0] = 0.01f * (gsums[0] / (float)N_PTS)
               + 0.01f * (gsums[1] / (float)(3 * N_PTS));
}

extern "C" void kernel_launch(void* const* d_in, const int* in_sizes, int n_in,
                              void* d_out, int out_size, void* d_ws, size_t ws_size,
                              hipStream_t stream) {
    const float* room_dims = (const float*)d_in[0];
    const float* coords    = (const float*)d_in[1];
    const float* time_raw  = (const float*)d_in[2];
    const float* W0 = (const float*)d_in[3];
    const float* b0 = (const float*)d_in[4];
    const float* W1 = (const float*)d_in[5];
    const float* b1 = (const float*)d_in[6];
    const float* W2 = (const float*)d_in[7];
    const float* b2 = (const float*)d_in[8];
    const float* W3 = (const float*)d_in[9];
    // d_in[10] = b3 (drops out of the Jacobian), d_in[11] = n_points

    float*    wsf   = (float*)d_ws;
    _Float16* wt1   = (_Float16*)((char*)d_ws + 256);
    _Float16* wt2   = wt1 + HID * HID;
    _Float16* wt3   = wt2 + HID * HID;
    float*    gsums = wsf + 4;

    (void)hipFuncSetAttribute(reinterpret_cast<const void*>(siren_pinn_kernel),
                              hipFuncAttributeMaxDynamicSharedMemorySize, SMEM_BYTES);

    hipLaunchKernelGGL(prep_kernel, dim3(1), dim3(32), 0, stream, room_dims, wsf);
    hipLaunchKernelGGL(wtrans_kernel, dim3(HID * HID / 256), dim3(256), 0, stream,
                       W1, W2, W3, wt1, wt2, wt3);
    hipLaunchKernelGGL(siren_pinn_kernel, dim3(N_PTS / PTSB), dim3(256), SMEM_BYTES, stream,
                       coords, time_raw, W0, b0, b1, b2, wt1, wt2, wt3, wsf, gsums);
    hipLaunchKernelGGL(loss_finish_kernel, dim3(1), dim3(1), 0, stream,
                       gsums, (float*)d_out);
}
